// DeepseekV2DecoderLayer_37271726195199
// MI455X (gfx1250) — matrix operand order repaired and checked by reading, not compiled, not run on hardware
//
#include <hip/hip_runtime.h>
#include <hip/hip_bf16.h>

typedef __attribute__((ext_vector_type(16))) __bf16 v16bf;
typedef __attribute__((ext_vector_type(8)))  float  v8f;

union FragU { uint4 q[2]; v16bf v; };

// ---------------------------------------------------------------------------
// fp32 -> bf16 conversions.
// Preferred: v_cvt_pk_bf16_f32 (if the toolchain exposes it).
// Fallback: round-half-up + v_perm_b32 byte-select -> 3 VALU per pair.
// ---------------------------------------------------------------------------
__device__ __forceinline__ unsigned pack_bf16(float a, float b) {
#if __has_builtin(__builtin_amdgcn_cvt_pk_bf16_f32)
  auto p = __builtin_amdgcn_cvt_pk_bf16_f32(a, b);
  union { decltype(p) v; unsigned u; } x; x.v = p;
  return x.u;
#else
  union { float f; unsigned u; } xa, xb;
  xa.f = a; xb.f = b;
  unsigned ua = xa.u + 0x8000u;   // round-half-up into bf16 mantissa
  unsigned ub = xb.u + 0x8000u;
  // result: low16 = ua[31:16] (bf16(a)), high16 = ub[31:16] (bf16(b))
  return __builtin_amdgcn_perm(ub, ua, 0x07060302u);
#endif
}

__device__ __forceinline__ unsigned short f2bf(float f) {
#if __has_builtin(__builtin_amdgcn_cvt_pk_bf16_f32)
  return (unsigned short)(pack_bf16(f, 0.0f) & 0xFFFFu);
#else
  union { float f; unsigned u; } x; x.f = f;
  return (unsigned short)((x.u + 0x8000u) >> 16);
#endif
}

__device__ __forceinline__ v8f wmma_bf16(const FragU& a, const FragU& b, v8f c) {
  return __builtin_amdgcn_wmma_f32_16x16x32_bf16(false, a.v, false, b.v,
                                                 (short)0, c, false, false);
}

// ---------------------------------------------------------------------------
// Generic bf16-WMMA GEMM: C[M,N] = A[M,K] * B[K,N]  (fp32 in, bf16 math,
// fp32 accumulate). transB=1: B given as [N,K]. Batch over blockIdx.z with
// element strides bBatch/cBatch. Output fp32 (Cf) or bf16 (Ch); cTrans=1
// writes Ch[col*ldc + row] (transposed bf16 output, used for V^T).
// Requires M%128==0, N%64==0, K%32==0 (true for every call below).
// ---------------------------------------------------------------------------
__global__ __launch_bounds__(256) void gemm_wmma(
    const float* __restrict__ A, int lda,
    const float* __restrict__ B, int transB, long long bBatch,
    float* __restrict__ Cf, unsigned short* __restrict__ Ch, int cTrans,
    int ldc, long long cBatch,
    int N, int K)
{
  __shared__ __attribute__((aligned(16))) unsigned short As[128][40];
  __shared__ __attribute__((aligned(16))) unsigned short Bs[64][40];
  const int tid  = threadIdx.x;
  const int wave = tid >> 5, lane = tid & 31, ln = lane & 15, lh = lane >> 4;
  const int m0 = blockIdx.y * 128, n0 = blockIdx.x * 64;
  B += (long long)blockIdx.z * bBatch;
  const long long cOff = (long long)blockIdx.z * cBatch;

  v8f acc[4];
  for (int t = 0; t < 4; ++t)
    for (int i = 0; i < 8; ++i) acc[t][i] = 0.0f;

  const int arow = tid >> 1, akp = (tid & 1) * 16;
  const int bn = tid >> 2,  bkp = (tid & 3) * 8;

  for (int k0 = 0; k0 < K; k0 += 32) {
    __syncthreads();
    { // stage A tile 128x32 (fp32 -> packed bf16)
      const float4* src = (const float4*)(A + (long long)(m0 + arow) * lda + k0 + akp);
      float4 f0 = src[0], f1 = src[1], f2 = src[2], f3 = src[3];
      uint4 w0, w1;
      w0.x = pack_bf16(f0.x, f0.y); w0.y = pack_bf16(f0.z, f0.w);
      w0.z = pack_bf16(f1.x, f1.y); w0.w = pack_bf16(f1.z, f1.w);
      w1.x = pack_bf16(f2.x, f2.y); w1.y = pack_bf16(f2.z, f2.w);
      w1.z = pack_bf16(f3.x, f3.y); w1.w = pack_bf16(f3.z, f3.w);
      *(uint4*)&As[arow][akp]     = w0;
      *(uint4*)&As[arow][akp + 8] = w1;
      if (k0 + 32 < K)  // prefetch next A tile (global_prefetch_b8)
        __builtin_prefetch(A + (long long)(m0 + arow) * lda + k0 + 32 + akp, 0, 0);
    }
    { // stage B tile as Bs[n][k]
      uint4 w;
      if (transB) {
        const float4* src = (const float4*)(B + (long long)(n0 + bn) * K + k0 + bkp);
        float4 f0 = src[0], f1 = src[1];
        w.x = pack_bf16(f0.x, f0.y); w.y = pack_bf16(f0.z, f0.w);
        w.z = pack_bf16(f1.x, f1.y); w.w = pack_bf16(f1.z, f1.w);
      } else {
        const float* src = B + (long long)(k0 + bkp) * N + (n0 + bn);
        float g[8];
        for (int i = 0; i < 8; ++i) g[i] = src[(long long)i * N];
        w.x = pack_bf16(g[0], g[1]); w.y = pack_bf16(g[2], g[3]);
        w.z = pack_bf16(g[4], g[5]); w.w = pack_bf16(g[6], g[7]);
      }
      *(uint4*)&Bs[bn][bkp] = w;
    }
    __syncthreads();

    FragU af;  // A operand: interleaved 8h K-chunks
    af.q[0] = *(const uint4*)&As[wave*16 + ln][8*lh];
    af.q[1] = *(const uint4*)&As[wave*16 + ln][16 + 8*lh];
    for (int t = 0; t < 4; ++t) {
      FragU bf;  // B operand: interleaved 8h K-chunks
      bf.q[0] = *(const uint4*)&Bs[t*16 + ln][8*lh];
      bf.q[1] = *(const uint4*)&Bs[t*16 + ln][16 + 8*lh];
      acc[t] = wmma_bf16(af, bf, acc[t]);
    }
  }

  for (int t = 0; t < 4; ++t)
    for (int r = 0; r < 8; ++r) {
      long long row = m0 + wave*16 + r + 8*lh;
      long long col = n0 + t*16 + ln;
      if (Cf)          Cf[cOff + row * (long long)ldc + col] = acc[t][r];
      else if (cTrans) Ch[cOff + col * (long long)ldc + row] = f2bf(acc[t][r]);
      else             Ch[cOff + row * (long long)ldc + col] = f2bf(acc[t][r]);
    }
}

// ---------------------------------------------------------------------------
// In-place RMSNorm over `len` elements of each row (row pitch `stride`).
// ---------------------------------------------------------------------------
__global__ __launch_bounds__(256) void rmsnorm_k(float* __restrict__ x,
                                                 const float* __restrict__ w,
                                                 int len, int stride)
{
  float* p = x + (long long)blockIdx.x * stride;
  __shared__ float red[256];
  float s = 0.f;
  for (int i = threadIdx.x; i < len; i += 256) { float v = p[i]; s += v * v; }
  red[threadIdx.x] = s;
  __syncthreads();
  for (int off = 128; off; off >>= 1) {
    if (threadIdx.x < off) red[threadIdx.x] += red[threadIdx.x + off];
    __syncthreads();
  }
  float scale = rsqrtf(red[0] / (float)len + 1e-6f);
  for (int i = threadIdx.x; i < len; i += 256) p[i] = p[i] * scale * w[i];
}

// ---------------------------------------------------------------------------
// k_pe: read kv[s, 512:576], interleaved->half + RoPE, broadcast bf16 into
// k_bf16[s, h, 128:192] for all 32 heads.  block = 64 threads.
// ---------------------------------------------------------------------------
__global__ void rope_kpe(const float* __restrict__ kv, const int* __restrict__ positions,
                         unsigned short* __restrict__ kbf)
{
  int s = blockIdx.x, j = threadIdx.x;
  __shared__ float x[64];
  x[j] = kv[(long long)s * 576 + 512 + j];
  __syncthreads();
  int pos = positions[s];
  int f = j & 31;
  float ang = (float)pos * __expf(-(2.0f * (float)f / 64.0f) * logf(10000.0f));
  float c = cosf(ang), sn = sinf(ang);
  float out = (j < 32) ? x[2*j] * c - x[2*j+1] * sn
                       : x[2*(j-32)+1] * c + x[2*(j-32)] * sn;
  unsigned short hb = f2bf(out);
  for (int h = 0; h < 32; ++h)
    kbf[((long long)s * 32 + h) * 192 + 128 + j] = hb;
}

// ---------------------------------------------------------------------------
// q_full: copy q_nope as bf16, RoPE the pe half. grid (2048,32), block 192.
// ---------------------------------------------------------------------------
__global__ void build_q(const float* __restrict__ qbig, const int* __restrict__ positions,
                        unsigned short* __restrict__ qbf)
{
  int s = blockIdx.x, h = blockIdx.y, t = threadIdx.x;
  const float* src = qbig + ((long long)s * 32 + h) * 192;
  unsigned short* dst = qbf + ((long long)s * 32 + h) * 192;
  __shared__ float x[64];
  if (t >= 128) x[t - 128] = src[t];
  __syncthreads();
  if (t < 128) {
    dst[t] = f2bf(src[t]);
  } else {
    int j = t - 128, f = j & 31;
    int pos = positions[s];
    float ang = (float)pos * __expf(-(2.0f * (float)f / 64.0f) * logf(10000.0f));
    float c = cosf(ang), sn = sinf(ang);
    float out = (j < 32) ? x[2*j] * c - x[2*j+1] * sn
                         : x[2*(j-32)+1] * c + x[2*(j-32)] * sn;
    dst[t] = f2bf(out);
  }
}

// ---------------------------------------------------------------------------
// Causal flash attention, one (64-query-row block, head) per workgroup.
// 4 waves x 16 rows; keys in blocks of 32; QK^T and P*V via bf16 WMMA.
// Q:[S][32][192] bf16, K:[S][32][192] bf16, VT:[32][128][S] bf16 (pre-
// transposed by the einsum GEMM), O:[S][4096] fp32.
// ---------------------------------------------------------------------------
__global__ __launch_bounds__(128) void mla_attn(
    const unsigned short* __restrict__ Q,
    const unsigned short* __restrict__ Kb,
    const unsigned short* __restrict__ VT,
    float* __restrict__ O)
{
  const int head = blockIdx.y, qblk = blockIdx.x, tid = threadIdx.x;
  const int wave = tid >> 5, lane = tid & 31, ln = lane & 15, lh = lane >> 4;
  const int qbase = qblk * 64 + wave * 16;

  __shared__ __attribute__((aligned(16))) unsigned short Vt[128][40];   // V^T tile
  __shared__ __attribute__((aligned(16))) unsigned short Pl[4][16][40]; // per-wave P

  FragU qf[6];  // Q rows resident in registers (A-operand layout)
  {
    const unsigned short* qrow = Q + ((long long)(qbase + ln) * 32 + head) * 192;
    for (int c = 0; c < 6; ++c) {
      qf[c].q[0] = *(const uint4*)(qrow + c*32 + 8*lh);
      qf[c].q[1] = *(const uint4*)(qrow + c*32 + 16 + 8*lh);
    }
  }

  v8f oacc[8];
  for (int t = 0; t < 8; ++t)
    for (int i = 0; i < 8; ++i) oacc[t][i] = 0.0f;
  float rmax[8], rsum[8];
  for (int r = 0; r < 8; ++r) { rmax[r] = -1e30f; rsum[r] = 0.0f; }
  const float scale = 0.07216878364870323f;  // 1/sqrt(192)

  const int kblocks = (qblk + 1) * 2;  // causal: keys <= qblk*64+63
  for (int kb = 0; kb < kblocks; ++kb) {
    const int kstart = kb * 32;
    __syncthreads();
    { // stage V^T tile: thread tid owns dv=tid, copies 32 contiguous keys
      const uint4* src = (const uint4*)(VT + ((long long)head * 128 + tid) * 2048 + kstart);
      uint4* dst = (uint4*)&Vt[tid][0];
      dst[0] = src[0]; dst[1] = src[1]; dst[2] = src[2]; dst[3] = src[3];
    }
    __syncthreads();

    // S = Q * K^T over 6 K-chunks, two 16-key column tiles
    v8f s0, s1;
    for (int i = 0; i < 8; ++i) { s0[i] = 0.0f; s1[i] = 0.0f; }
    for (int c = 0; c < 6; ++c) {
      const unsigned short* kr0 = Kb + ((long long)(kstart + ln) * 32 + head) * 192 + c*32;
      FragU kf0; kf0.q[0] = *(const uint4*)(kr0 + 8*lh); kf0.q[1] = *(const uint4*)(kr0 + 16 + 8*lh);
      s0 = wmma_bf16(qf[c], kf0, s0);
      const unsigned short* kr1 = Kb + ((long long)(kstart + 16 + ln) * 32 + head) * 192 + c*32;
      FragU kf1; kf1.q[0] = *(const uint4*)(kr1 + 8*lh); kf1.q[1] = *(const uint4*)(kr1 + 16 + 8*lh);
      s1 = wmma_bf16(qf[c], kf1, s1);
    }

    // online softmax (row = qbase + r + 8*lh, cols across 16-lane group)
    float p0[8], p1[8], alpha[8];
    for (int r = 0; r < 8; ++r) {
      int row = qbase + r + 8*lh;
      float v0 = s0[r] * scale; if (kstart + ln > row)      v0 = -1e30f;
      float v1 = s1[r] * scale; if (kstart + 16 + ln > row) v1 = -1e30f;
      float mx = fmaxf(v0, v1);
      mx = fmaxf(mx, __shfl_xor(mx, 1, 32));
      mx = fmaxf(mx, __shfl_xor(mx, 2, 32));
      mx = fmaxf(mx, __shfl_xor(mx, 4, 32));
      mx = fmaxf(mx, __shfl_xor(mx, 8, 32));
      float nm = fmaxf(rmax[r], mx);
      float a  = __expf(rmax[r] - nm);
      float e0 = __expf(v0 - nm), e1 = __expf(v1 - nm);
      float sm = e0 + e1;
      sm += __shfl_xor(sm, 1, 32);
      sm += __shfl_xor(sm, 2, 32);
      sm += __shfl_xor(sm, 4, 32);
      sm += __shfl_xor(sm, 8, 32);
      rsum[r] = rsum[r] * a + sm;
      rmax[r] = nm;
      alpha[r] = a; p0[r] = e0; p1[r] = e1;
    }
    for (int t = 0; t < 8; ++t)
      for (int r = 0; r < 8; ++r) oacc[t][r] *= alpha[r];

    // transpose P (C layout -> A layout) through per-wave LDS, then P*V
    for (int r = 0; r < 8; ++r) {
      Pl[wave][r + 8*lh][ln]      = f2bf(p0[r]);
      Pl[wave][r + 8*lh][16 + ln] = f2bf(p1[r]);
    }
    FragU pf;
    pf.q[0] = *(const uint4*)&Pl[wave][ln][8*lh];
    pf.q[1] = *(const uint4*)&Pl[wave][ln][16 + 8*lh];
    for (int t = 0; t < 8; ++t) {
      FragU vf;
      vf.q[0] = *(const uint4*)&Vt[t*16 + ln][8*lh];
      vf.q[1] = *(const uint4*)&Vt[t*16 + ln][16 + 8*lh];
      oacc[t] = wmma_bf16(pf, vf, oacc[t]);
    }
  }

  for (int t = 0; t < 8; ++t)
    for (int r = 0; r < 8; ++r) {
      long long row = qbase + r + 8*lh;
      O[row * 4096 + head * 128 + t*16 + ln] = oacc[t][r] / rsum[r];
    }
}

// ---------------------------------------------------------------------------
extern "C" void kernel_launch(void* const* d_in, const int* in_sizes, int n_in,
                              void* d_out, int out_size, void* d_ws, size_t ws_size,
                              hipStream_t stream) {
  const float* hidden    = (const float*)d_in[0];
  const int*   positions = (const int*)  d_in[1];
  const float* w_qa      = (const float*)d_in[2];
  const float* q_a_ln_w  = (const float*)d_in[3];
  const float* w_qb      = (const float*)d_in[4];
  const float* w_kva     = (const float*)d_in[5];
  const float* kv_a_ln_w = (const float*)d_in[6];
  const float* kc        = (const float*)d_in[7];
  const float* vc        = (const float*)d_in[8];
  const float* w_o       = (const float*)d_in[9];
  float* out = (float*)d_out;

  char* p = (char*)d_ws;
  float* q_a  = (float*)p;           p += (size_t)2048*1536*4;   // q lora (pre/post norm)
  float* kvb  = (float*)p;           p += (size_t)2048*576*4;    // kv (+ ckv in-place)
  float* qbig = (float*)p;           p += (size_t)2048*6144*4;   // q projection fp32
  unsigned short* qbf = (unsigned short*)p; p += (size_t)2048*32*192*2; // q_full bf16
  unsigned short* kbf = (unsigned short*)p; p += (size_t)2048*32*192*2; // k_full bf16
  unsigned short* vbf = (unsigned short*)p; p += (size_t)32*128*2048*2; // V^T bf16 [h][dv][s]
  float* obuf = (float*)p;           p += (size_t)2048*4096*4;   // attn out fp32

  // 1) q_a = hidden @ w_qa            [2048x5120]x[5120x1536]
  gemm_wmma<<<dim3(24,16,1),256,0,stream>>>(hidden,5120, w_qa,0,0,
                                            q_a,nullptr,0,1536,0, 1536,5120);
  // 2) kv = hidden @ w_kva            [2048x5120]x[5120x576]
  gemm_wmma<<<dim3(9,16,1),256,0,stream>>>(hidden,5120, w_kva,0,0,
                                           kvb,nullptr,0,576,0, 576,5120);
  // 3) RoPE k_pe -> broadcast into kbf[:,:,128:192] (reads cols 512..575 only)
  rope_kpe<<<dim3(2048),64,0,stream>>>(kvb, positions, kbf);
  // 4) RMSNorms (in place)
  rmsnorm_k<<<dim3(2048),256,0,stream>>>(q_a, q_a_ln_w, 1536, 1536);
  rmsnorm_k<<<dim3(2048),256,0,stream>>>(kvb, kv_a_ln_w, 512, 576);
  // 5) q = q_lora @ w_qb              [2048x1536]x[1536x6144]
  gemm_wmma<<<dim3(96,16,1),256,0,stream>>>(q_a,1536, w_qb,0,0,
                                            qbig,nullptr,0,6144,0, 6144,1536);
  build_q<<<dim3(2048,32),192,0,stream>>>(qbig, positions, qbf);
  // 6) k_nope[h] = ckv @ kc[h]^T  -> kbf[:,h,0:128]   (kc[h] is [128,512] = [N,K])
  gemm_wmma<<<dim3(2,16,32),256,0,stream>>>(kvb,576, kc,1,(long long)128*512,
                                            nullptr,kbf,0,6144,192, 128,512);
  // 7) v[h] = ckv @ vc[h] -> V^T layout vbf[h][dv][s]  (vc[h] is [512,128] = [K,N])
  gemm_wmma<<<dim3(2,16,32),256,0,stream>>>(kvb,576, vc,0,(long long)512*128,
                                            nullptr,vbf,1,2048,(long long)128*2048, 128,512);
  // 8) causal flash attention -> obuf [2048 x 4096]
  mla_attn<<<dim3(32,32),128,0,stream>>>(qbf, kbf, vbf, obuf);
  // 9) out = obuf @ w_o               [2048x4096]x[4096x5120]
  gemm_wmma<<<dim3(80,16,1),256,0,stream>>>(obuf,4096, w_o,0,0,
                                            out,nullptr,0,5120,0, 5120,4096);
}
